// JetLayer_37804302139697
// MI455X (gfx1250) — matrix operand order repaired and checked by reading, not compiled, not run on hardware
//
#include <hip/hip_runtime.h>
#include <hip/hip_bf16.h>
#include <math.h>

#define N 1024
#define BATCH 32

typedef _Float16 h16;
typedef _Float16 v16h __attribute__((ext_vector_type(16)));
typedef _Float16 v8h  __attribute__((ext_vector_type(8)));
typedef float    v8f  __attribute__((ext_vector_type(8)));
typedef int      v4i_vs __attribute__((vector_size(16)));   // matches builtin param type

// gfx1250 async global->LDS copy path (ASYNCcnt-tracked, bypasses VGPRs).
#if __has_builtin(__builtin_amdgcn_global_load_async_to_lds_b128) && \
    __has_builtin(__builtin_amdgcn_s_wait_asynccnt)
#define USE_ASYNC_LDS 1
#else
#define USE_ASYNC_LDS 0
#endif

__device__ __forceinline__ void copy_b128_to_lds(const h16* g, h16* l) {
#if USE_ASYNC_LDS
  // (global int4*, lds int4*, imm offset, imm cpol) -> global_load_async_to_lds_b128
  __builtin_amdgcn_global_load_async_to_lds_b128(
      (__attribute__((address_space(1))) v4i_vs*)g,
      (__attribute__((address_space(3))) v4i_vs*)l, 0, 0);
#else
  *(v8h*)l = *(const v8h*)g;
#endif
}

__device__ __forceinline__ void wait_lds_copies() {
#if USE_ASYNC_LDS
  __builtin_amdgcn_s_wait_asynccnt(0);
#endif
}

__device__ __forceinline__ float wrap_dphi(float dp) {
  const float PI_F    = 3.14159265358979323846f;
  const float TWO_PI  = 6.28318530717958647692f;
  const float INV_2PI = 0.15915494309189533577f;
  float t = dp + PI_F;
  t -= floorf(t * INV_2PI) * TWO_PI;   // python-style mod: result in [0, 2pi)
  return t - PI_F;
}

// ---------------------------------------------------------------------------
// Zero the per-batch accumulators: acc[b*2+0] = raw ECF2, acc[b*2+1] = raw tr(W^3)
// ---------------------------------------------------------------------------
__global__ void zero_acc_kernel(float* __restrict__ acc) {
  int t = threadIdx.x;
  if (t < 2 * BATCH) acc[t] = 0.0f;
}

// ---------------------------------------------------------------------------
// Build W_ij = sqrt(pt_i) * sqrt(pt_j) * R_ij  (f16, row-major NxN per batch),
// fused raw-ECF2 accumulation: sum_ij pt_i pt_j R_ij.
// Grid: (N/64, N/64, nb), block 256.
// ---------------------------------------------------------------------------
__global__ void build_w_kernel(const float* __restrict__ x, h16* __restrict__ W,
                               float* __restrict__ acc, int b0) {
  const int lb = blockIdx.z;
  const int gb = b0 + lb;
  const float* xb = x + (size_t)gb * N * 3;
  h16* Wb = W + (size_t)lb * N * N;
  const int i0 = blockIdx.x * 64;
  const int j0 = blockIdx.y * 64;

  __shared__ float si[64], ei[64], fi[64];
  __shared__ float sj[64], ej[64], fj[64];
  __shared__ float red[256];

  const int t = threadIdx.x;
  if (t < 64) {
    const float* p = xb + (size_t)(i0 + t) * 3;
    si[t] = sqrtf(p[0]); ei[t] = p[1]; fi[t] = p[2];
  } else if (t < 128) {
    int u = t - 64;
    const float* p = xb + (size_t)(j0 + u) * 3;
    sj[u] = sqrtf(p[0]); ej[u] = p[1]; fj[u] = p[2];
  }
  __syncthreads();

  float e2 = 0.0f;
  // 64x64 tile = 4096 halves; each thread produces two contiguous v8h stores
  #pragma unroll
  for (int q = 0; q < 2; ++q) {
    int cidx = t + q * 256;           // 0..511 chunks of 8 halves
    int li = cidx >> 3;               // row in tile
    int jc = (cidx & 7) * 8;          // col start in tile
    int gi = i0 + li;
    v8h wv;
    #pragma unroll
    for (int u = 0; u < 8; ++u) {
      int lj = jc + u;
      int gj = j0 + lj;
      float de = ei[li] - ej[lj];
      float dp = wrap_dphi(fi[li] - fj[lj]);
      float a = (gi == gj) ? 0.0f : sqrtf(de * de + dp * dp);  // A_ii = 0
      float ss = si[li] * sj[lj];
      float w = ss * a;
      wv[u] = (h16)w;
      e2 += ss * w;                    // = pt_i pt_j R_ij
    }
    *(v8h*)&Wb[(size_t)gi * N + j0 + jc] = wv;
  }

  red[t] = e2;
  __syncthreads();
  for (int s = 128; s > 0; s >>= 1) {
    if (t < s) red[t] += red[t + s];
    __syncthreads();
  }
  if (t == 0) atomicAdd(&acc[gb * 2 + 0], red[0]);
}

// ---------------------------------------------------------------------------
// tr(W^3) via WMMA: for each 128x128 (i,k) tile, T = W[i,:] @ W[:,k] accumulated
// with v_wmma_f32_16x16x32_f16, then partial += sum T_ik * W_ik.
// Grid: (N/128, N/128, nb), block 256 (8 waves, wave32).
// ---------------------------------------------------------------------------
#define TILE 128
#define KSTEP 32
#define A_STRIDE 40    // 32 + 8 halves pad
#define B_STRIDE 136   // 128 + 8 halves pad
#define A_HALVES (TILE * A_STRIDE)    // 5120
#define SMEM_HALVES (TILE * B_STRIDE) // 17408 (also holds epilogue W_ik tile)

__global__ void __launch_bounds__(256)
gemm_tr3_kernel(const h16* __restrict__ W, float* __restrict__ acc, int b0) {
  const int lb = blockIdx.z;
  const int gb = b0 + lb;
  const h16* Wb = W + (size_t)lb * N * N;
  const int i0 = blockIdx.x * TILE;
  const int k0 = blockIdx.y * TILE;

  __shared__ h16 smem[SMEM_HALVES];  // K-loop: As | Bs ; epilogue: full W_ik tile
  __shared__ float red[256];
  h16* As = smem;                    // TILE x KSTEP (stride A_STRIDE)
  h16* Bs = smem + A_HALVES;         // KSTEP x TILE (stride B_STRIDE)

  const int t = threadIdx.x;
  const int lane  = t & 31;
  const int wave  = t >> 5;
  const int wm    = wave & 1;    // 0..1 : 64-row block
  const int wn    = wave >> 1;   // 0..3 : 32-col block
  const int lhalf = lane >> 4;   // lane group 0/1
  const int lmod  = lane & 15;

  v8f c[4][2] = {};                // 8 f32 16x16 accumulators per wave

  for (int kk = 0; kk < N; kk += KSTEP) {
    __syncthreads();
    // Stage A: W[i0..i0+127, kk..kk+31]; Stage B: W[kk..kk+31, k0..k0+127]
    #pragma unroll
    for (int q = 0; q < 2; ++q) {
      int cidx = t + q * 256;                       // 512 chunks of 8 halves each
      int arow = cidx >> 2, acc_ = cidx & 3;
      copy_b128_to_lds(&Wb[(size_t)(i0 + arow) * N + kk + acc_ * 8],
                       &As[arow * A_STRIDE + acc_ * 8]);
      int brow = cidx >> 4, bcc = cidx & 15;
      copy_b128_to_lds(&Wb[(size_t)(kk + brow) * N + k0 + bcc * 8],
                       &Bs[brow * B_STRIDE + bcc * 8]);
    }
    wait_lds_copies();
    __syncthreads();

    // A fragments (16x32 f16): lane m = lane%16; K-chunks split by lane half
    v16h a[4];
    #pragma unroll
    for (int fm = 0; fm < 4; ++fm) {
      int row = wm * 64 + fm * 16 + lmod;
      union { v16h v; v8h h[2]; } u;
      u.h[0] = *(const v8h*)&As[row * A_STRIDE + lhalf * 8];        // K 0..7 / 8..15
      u.h[1] = *(const v8h*)&As[row * A_STRIDE + 16 + lhalf * 8];   // K 16..23 / 24..31
      a[fm] = u.v;
    }
    // B fragments (32x16 f16): lane n = lane%16; K order matches A (lane half selects K chunks)
    v16h b[2];
    #pragma unroll
    for (int fn = 0; fn < 2; ++fn) {
      int col = wn * 32 + fn * 16;
      v16h v;
      #pragma unroll
      for (int s = 0; s < 16; ++s) {
        int k = (s < 8) ? (lhalf * 8 + s) : (16 + lhalf * 8 + (s - 8));
        v[s] = Bs[k * B_STRIDE + col + lmod];
      }
      b[fn] = v;
    }

    #pragma unroll
    for (int fm = 0; fm < 4; ++fm)
      #pragma unroll
      for (int fn = 0; fn < 2; ++fn)
        c[fm][fn] = __builtin_amdgcn_wmma_f32_16x16x32_f16(
            false, a[fm], false, b[fn], (short)0, c[fm][fn], false, false);
  }

  // ---- Epilogue: stage W_ik (128x128) into LDS, then partial += sum T .* W ----
  __syncthreads();
  #pragma unroll
  for (int q = 0; q < 8; ++q) {
    int cidx = t + q * 256;          // 2048 chunks of 8 halves
    int row = cidx >> 4, cc = cidx & 15;
    copy_b128_to_lds(&Wb[(size_t)(i0 + row) * N + k0 + cc * 8],
                     &smem[row * B_STRIDE + cc * 8]);
  }
  wait_lds_copies();
  __syncthreads();

  // C-layout: VGPR r, lanes 0-15: M=r, N=lane; lanes 16-31: M=8+r, N=lane-16
  float part = 0.0f;
  #pragma unroll
  for (int fm = 0; fm < 4; ++fm) {
    #pragma unroll
    for (int fn = 0; fn < 2; ++fn) {
      #pragma unroll
      for (int r = 0; r < 8; ++r) {
        int m = wm * 64 + fm * 16 + r + 8 * lhalf;
        int n = wn * 32 + fn * 16 + lmod;
        float wik = (float)smem[m * B_STRIDE + n];
        part += c[fm][fn][r] * wik;
      }
    }
  }
  red[t] = part;
  __syncthreads();
  for (int s = 128; s > 0; s >>= 1) {
    if (t < s) red[t] += red[t + s];
    __syncthreads();
  }
  if (t == 0) atomicAdd(&acc[gb * 2 + 1], red[0]);
}

// ---------------------------------------------------------------------------
// Per-batch kinematics + combine ECFs into outputs [B,6].
// ---------------------------------------------------------------------------
__global__ void finalize_kernel(const float* __restrict__ x,
                                const float* __restrict__ acc,
                                float* __restrict__ out) {
  const int b = blockIdx.x;
  const float* xb = x + (size_t)b * N * 3;
  __shared__ float red[5][256];
  const int t = threadIdx.x;

  float spt = 0, spx = 0, spy = 0, spz = 0, se = 0;
  for (int i = t; i < N; i += 256) {
    float pt = xb[i * 3 + 0];
    float eta = xb[i * 3 + 1];
    float phi = xb[i * 3 + 2];
    spt += pt;
    spx += pt * cosf(phi);
    spy += pt * sinf(phi);
    spz += pt * sinhf(eta);
    se  += pt * coshf(eta);
  }
  red[0][t] = spt; red[1][t] = spx; red[2][t] = spy; red[3][t] = spz; red[4][t] = se;
  __syncthreads();
  for (int s = 128; s > 0; s >>= 1) {
    if (t < s) {
      #pragma unroll
      for (int k = 0; k < 5; ++k) red[k][t] += red[k][t + s];
    }
    __syncthreads();
  }
  if (t == 0) {
    float ecf1 = red[0][0];
    float px = red[1][0], py = red[2][0], pz = red[3][0], e = red[4][0];
    float ecf2 = 0.5f * acc[b * 2 + 0];
    float ecf3 = (1.0f / 6.0f) * acc[b * 2 + 1];
    float c2 = ecf3 * ecf1 / (ecf2 * ecf2);
    float d2 = ecf3 * ecf1 * ecf1 * ecf1 / (ecf2 * ecf2 * ecf2);
    float jpt  = sqrtf(px * px + py * py);
    float jeta = asinhf(pz / fmaxf(jpt, 1e-12f));
    float jphi = atan2f(py, px);
    float m2   = e * e - (px * px + py * py + pz * pz);
    float jm   = sqrtf(fmaxf(m2, 1e-12f));
    out[b * 6 + 0] = jpt;
    out[b * 6 + 1] = jeta;
    out[b * 6 + 2] = jphi;
    out[b * 6 + 3] = jm;
    out[b * 6 + 4] = c2;
    out[b * 6 + 5] = d2;
  }
}

// ---------------------------------------------------------------------------
// Launch: zero accumulators; per workspace chunk: build W (f16) then WMMA
// tr(W^3); finally combine. All stream-ordered, graph-capture safe.
// ws layout: [0,256): float acc[32][2]; [1024, ...): f16 W chunks (2MB/batch).
// ---------------------------------------------------------------------------
extern "C" void kernel_launch(void* const* d_in, const int* in_sizes, int n_in,
                              void* d_out, int out_size, void* d_ws, size_t ws_size,
                              hipStream_t stream) {
  (void)in_sizes; (void)n_in; (void)out_size;
  const float* x = (const float*)d_in[0];
  float* out = (float*)d_out;
  float* acc = (float*)d_ws;
  h16* W = (h16*)((char*)d_ws + 1024);

  const size_t per_batch = (size_t)N * N * sizeof(h16);   // 2 MB
  size_t avail = ws_size > 1024 ? ws_size - 1024 : 0;
  int maxb = (int)(avail / per_batch);
  if (maxb < 1) maxb = 1;          // minimal assumption: ws holds one W matrix
  if (maxb > BATCH) maxb = BATCH;  // 64 MB -> single chunk, W fully L2-resident

  zero_acc_kernel<<<1, 64, 0, stream>>>(acc);
  for (int b0 = 0; b0 < BATCH; b0 += maxb) {
    int nb = BATCH - b0;
    if (nb > maxb) nb = maxb;
    build_w_kernel<<<dim3(N / 64, N / 64, nb), 256, 0, stream>>>(x, W, acc, b0);
    gemm_tr3_kernel<<<dim3(N / TILE, N / TILE, nb), 256, 0, stream>>>(W, acc, b0);
  }
  finalize_kernel<<<BATCH, 256, 0, stream>>>(x, acc, out);
}
